// EdgeBlockSum_84104049590406
// MI455X (gfx1250) — compile-verified, matrix operand order checked by reading, not run on hardware
//
#include <hip/hip_runtime.h>

// ---------------------------------------------------------------------------
// EdgeBlockSum on MI455X (gfx1250, wave32, WMMA f32_16x16x32_f16)
//
//  k0: convert the four 128x128 f32 weight matrices to f16 (once, into ws)
//  k1: node projections  sproj = nfeat @ Ws^T, dproj = nfeat @ Wd^T  (WMMA)
//  k2: per 16-edge tile: efeat@We^T (WMMA) + gather(sproj,dproj) + b1
//      -> SiLU -> LDS transpose -> h@Wo^T (WMMA) + b_o -> LayerNorm
//      -> + efeat residual -> store
//
//  L2 policy: streaming output writes are NON-TEMPORAL so the 51 MB of
//  gather tables (sproj/dproj) stay resident in the 192 MB L2.
//  Stage-0 efeat tile staging uses GLOBAL_LOAD_ASYNC_TO_LDS_B128
//  (ASYNCcnt-tracked, direct memory->LDS, no VGPR round trip).
// ---------------------------------------------------------------------------

typedef __attribute__((ext_vector_type(16))) _Float16 v16h;
typedef __attribute__((ext_vector_type(8)))  _Float16 v8h;
typedef __attribute__((ext_vector_type(8)))  float    v8f;
typedef __attribute__((ext_vector_type(4)))  float    v4f;
typedef __attribute__((ext_vector_type(4)))  int      v4i;

typedef __attribute__((address_space(1))) v4i* gv4i_p;   // global int4*
typedef __attribute__((address_space(3))) v4i* lv4i_p;   // LDS int4*

#define DIM      128
#define N_NODES  50000
#define N_EDGES  600000
#define WV       4            // waves per block in the edge kernel
static constexpr float LN_EPS = 1e-5f;

#if defined(__has_builtin)
#if __has_builtin(__builtin_amdgcn_global_load_async_to_lds_b128)
#define HAVE_ASYNC_LDS 1
#endif
#endif

// A/B operand for v_wmma_f32_16x16x32_f16, built from an f32 row (cvt to f16).
// Lane layout: elems 0..7 = K base..base+7 ; elems 8..15 = K base+16..base+23
__device__ __forceinline__ v16h load_ab_f32(const float* __restrict__ row, int base) {
  v4f f0 = *(const v4f*)(row + base);
  v4f f1 = *(const v4f*)(row + base + 4);
  v4f f2 = *(const v4f*)(row + base + 16);
  v4f f3 = *(const v4f*)(row + base + 20);
  v16h a;
#pragma unroll
  for (int i = 0; i < 4; ++i) {
    a[i]      = (_Float16)f0[i];
    a[4 + i]  = (_Float16)f1[i];
    a[8 + i]  = (_Float16)f2[i];
    a[12 + i] = (_Float16)f3[i];
  }
  return a;
}

// Same operand from an f16 row (weights pre-converted, or LDS-staged h).
__device__ __forceinline__ v16h load_ab_f16(const _Float16* __restrict__ row, int base) {
  v8h b0 = *(const v8h*)(row + base);
  v8h b1 = *(const v8h*)(row + base + 16);
  v16h b;
#pragma unroll
  for (int i = 0; i < 8; ++i) { b[i] = b0[i]; b[8 + i] = b1[i]; }
  return b;
}

// --------------------------- k0: weight conversion --------------------------
__global__ void convert_weights_kernel(const float* __restrict__ We,
                                       const float* __restrict__ Ws,
                                       const float* __restrict__ Wd,
                                       const float* __restrict__ Wo,
                                       _Float16* __restrict__ out) {
  int i = blockIdx.x * blockDim.x + threadIdx.x;
  if (i < DIM * DIM) {
    out[i]                 = (_Float16)We[i];
    out[DIM * DIM + i]     = (_Float16)Ws[i];
    out[2 * DIM * DIM + i] = (_Float16)Wd[i];
    out[3 * DIM * DIM + i] = (_Float16)Wo[i];
  }
}

// --------------------------- k1: node projections ---------------------------
// grid = (N_NODES/16, 8), block = 64 (2 waves: wave0 -> sproj, wave1 -> dproj)
__global__ __launch_bounds__(64) void node_proj_kernel(
    const float* __restrict__ nfeat,
    const _Float16* __restrict__ Wsh, const _Float16* __restrict__ Wdh,
    float* __restrict__ sproj, float* __restrict__ dproj) {
  const int lane = threadIdx.x & 31;
  const int wv   = threadIdx.x >> 5;      // 0 = src-proj, 1 = dst-proj
  const int lo   = lane & 15, hi = lane >> 4;
  const int tile = blockIdx.x;            // 16-node row tile
  const int nt   = blockIdx.y;            // 16-col output tile

  const _Float16* W  = wv ? Wdh : Wsh;
  float* out         = wv ? dproj : sproj;
  const float*    aRow = nfeat + (size_t)(tile * 16 + lo) * DIM;
  const _Float16* bRow = W + (size_t)(nt * 16 + lo) * DIM;

  v8f acc = {};
#pragma unroll
  for (int kc = 0; kc < 4; ++kc) {
    const int base = kc * 32 + hi * 8;
    v16h a = load_ab_f32(aRow, base);
    v16h b = load_ab_f16(bRow, base);
    acc = __builtin_amdgcn_wmma_f32_16x16x32_f16(false, a, false, b,
                                                 (short)0, acc, false, false);
  }
#pragma unroll
  for (int j = 0; j < 8; ++j) {
    const int m = j + 8 * hi;
    out[(size_t)(tile * 16 + m) * DIM + nt * 16 + lo] = acc[j];
  }
}

// ------------------------------ k2: edge kernel -----------------------------
// One wave per 16-edge tile; block = 4 independent waves.
__global__ __launch_bounds__(32 * WV) void edge_mlp_kernel(
    const float* __restrict__ efeat,
    const int* __restrict__ src, const int* __restrict__ dst,
    const _Float16* __restrict__ Weh, const _Float16* __restrict__ Woh,
    const float* __restrict__ sproj, const float* __restrict__ dproj,
    const float* __restrict__ b1, const float* __restrict__ b_o,
    const float* __restrict__ gamma, const float* __restrict__ beta,
    float* __restrict__ out) {
  __shared__ __align__(16) float    ef_s[WV][16][DIM];   // 32 KB: f32 efeat tiles
  __shared__ __align__(16) _Float16 h_s [WV][16][DIM];   // 16 KB: f16 hidden tiles

  const int  lane = threadIdx.x & 31;
  const int  wv   = threadIdx.x >> 5;
  const int  lo   = lane & 15, hi = lane >> 4;
  const long tile = (long)blockIdx.x * WV + wv;
  const long e0   = tile * 16;

  // ---- stage 0: efeat tile -> LDS (f32, reused for GEMM-A and residual)
#ifdef HAVE_ASYNC_LDS
  {
    // direct memory->LDS DMA, 16 B per lane per op, ASYNCcnt-tracked
    const char* gbase = (const char*)(efeat + e0 * DIM);
    char*       lbase = (char*)&ef_s[wv][0][0];
#pragma unroll
    for (int r = 0; r < 16; ++r) {
      const size_t off = ((size_t)r * DIM + lane * 4) * sizeof(float);
      __builtin_amdgcn_global_load_async_to_lds_b128(
          (gv4i_p)(gbase + off), (lv4i_p)(lbase + off), 0, 0);
    }
  }
#if __has_builtin(__builtin_amdgcn_s_wait_asynccnt)
  __builtin_amdgcn_s_wait_asynccnt(0);
#else
  asm volatile("s_wait_asynccnt 0" ::: "memory");
#endif
  asm volatile("" ::: "memory");
#else
#pragma unroll
  for (int r = 0; r < 16; ++r) {
    // non-temporal: efeat is streamed once, keep it out of L2 residency
    v4f v = __builtin_nontemporal_load((const v4f*)(efeat + (e0 + r) * DIM + lane * 4));
    *(v4f*)&ef_s[wv][r][lane * 4] = v;
  }
  asm volatile("s_wait_dscnt 0" ::: "memory");
#endif

  // ---- stage 1: acc[nt] = efeat_tile @ We^T  (16x128 out, f32 accum)
  v8f acc[8];
#pragma unroll
  for (int nt = 0; nt < 8; ++nt) acc[nt] = (v8f){};
  {
    const float* aRow = &ef_s[wv][lo][0];
#pragma unroll
    for (int kc = 0; kc < 4; ++kc) {
      const int base = kc * 32 + hi * 8;
      v16h a = load_ab_f32(aRow, base);
#pragma unroll
      for (int nt = 0; nt < 8; ++nt) {
        v16h b = load_ab_f16(Weh + (size_t)(nt * 16 + lo) * DIM, base);
        acc[nt] = __builtin_amdgcn_wmma_f32_16x16x32_f16(false, a, false, b,
                                                         (short)0, acc[nt], false, false);
      }
    }
  }

  // ---- stage 2: + sproj[src] + dproj[dst] + b1, SiLU, -> LDS as f16 A-tile
  float b1v[8];
#pragma unroll
  for (int nt = 0; nt < 8; ++nt) b1v[nt] = b1[nt * 16 + lo];
#pragma unroll
  for (int j = 0; j < 8; ++j) {
    const int  m = j + 8 * hi;
    const long e = e0 + m;
    const float* sp = sproj + (size_t)src[e] * DIM;   // L2-resident gathers
    const float* dp = dproj + (size_t)dst[e] * DIM;
#pragma unroll
    for (int nt = 0; nt < 8; ++nt) {
      const int n = nt * 16 + lo;
      float x = acc[nt][j] + sp[n] + dp[n] + b1v[nt];
      x = x / (1.0f + __expf(-x));            // SiLU
      h_s[wv][m][n] = (_Float16)x;
    }
  }
  asm volatile("s_wait_dscnt 0" ::: "memory");

  // ---- stage 3: acc2[nt] = h_tile @ Wo^T  (A read straight from LDS as f16)
  v8f acc2[8];
#pragma unroll
  for (int nt = 0; nt < 8; ++nt) acc2[nt] = (v8f){};
  {
    const _Float16* aRow = &h_s[wv][lo][0];
#pragma unroll
    for (int kc = 0; kc < 4; ++kc) {
      const int base = kc * 32 + hi * 8;
      v16h a = load_ab_f16(aRow, base);
#pragma unroll
      for (int nt = 0; nt < 8; ++nt) {
        v16h b = load_ab_f16(Woh + (size_t)(nt * 16 + lo) * DIM, base);
        acc2[nt] = __builtin_amdgcn_wmma_f32_16x16x32_f16(false, a, false, b,
                                                          (short)0, acc2[nt], false, false);
      }
    }
  }

  // ---- stage 4: + b_o, LayerNorm over 128, + residual, store (non-temporal)
  float g[8], bt[8], bo[8];
#pragma unroll
  for (int nt = 0; nt < 8; ++nt) {
    const int n = nt * 16 + lo;
    g[nt] = gamma[n]; bt[nt] = beta[n]; bo[nt] = b_o[n];
  }
#pragma unroll
  for (int j = 0; j < 8; ++j) {
    const int m = j + 8 * hi;
    float vals[8], s = 0.f, q = 0.f;
#pragma unroll
    for (int nt = 0; nt < 8; ++nt) {
      float v = acc2[nt][j] + bo[nt];
      vals[nt] = v; s += v; q += v * v;
    }
    // reduce across the 16 lanes (lo) that share this row
#pragma unroll
    for (int msk = 1; msk < 16; msk <<= 1) {
      s += __shfl_xor(s, msk, 32);
      q += __shfl_xor(q, msk, 32);
    }
    const float mu  = s * (1.0f / DIM);
    const float var = q * (1.0f / DIM) - mu * mu;
    const float inv = rsqrtf(var + LN_EPS);
    float* orow = out + (size_t)(e0 + m) * DIM;
#pragma unroll
    for (int nt = 0; nt < 8; ++nt) {
      const int n = nt * 16 + lo;
      const float y = (vals[nt] - mu) * inv * g[nt] + bt[nt] + ef_s[wv][m][n];
      __builtin_nontemporal_store(y, &orow[n]);   // streamed output: TH=NT
    }
  }
}

// ------------------------------------------------------------------ launcher
extern "C" void kernel_launch(void* const* d_in, const int* in_sizes, int n_in,
                              void* d_out, int out_size, void* d_ws, size_t ws_size,
                              hipStream_t stream) {
  const float* efeat = (const float*)d_in[0];
  const float* nfeat = (const float*)d_in[1];
  const int*   src   = (const int*)d_in[2];
  const int*   dst   = (const int*)d_in[3];
  const float* W_e   = (const float*)d_in[4];
  const float* W_s   = (const float*)d_in[5];
  const float* W_d   = (const float*)d_in[6];
  const float* b1    = (const float*)d_in[7];
  const float* W_o   = (const float*)d_in[8];
  const float* b_o   = (const float*)d_in[9];
  const float* gamma = (const float*)d_in[10];
  const float* beta  = (const float*)d_in[11];

  float* out_e = (float*)d_out;                       // [600000,128]
  float* out_n = out_e + (size_t)N_EDGES * DIM;       // [50000,128] passthrough

  // workspace: sproj | dproj | 4x f16 weights (51.2 MB + 128 KB)
  float*    sproj = (float*)d_ws;
  float*    dproj = sproj + (size_t)N_NODES * DIM;
  _Float16* Wh    = (_Float16*)(dproj + (size_t)N_NODES * DIM);
  _Float16* Weh = Wh;
  _Float16* Wsh = Wh + DIM * DIM;
  _Float16* Wdh = Wh + 2 * DIM * DIM;
  _Float16* Woh = Wh + 3 * DIM * DIM;

  convert_weights_kernel<<<(DIM * DIM + 255) / 256, 256, 0, stream>>>(W_e, W_s, W_d, W_o, Wh);

  node_proj_kernel<<<dim3(N_NODES / 16, 8), 64, 0, stream>>>(nfeat, Wsh, Wdh, sproj, dproj);

  edge_mlp_kernel<<<(N_EDGES / 16) / WV, 32 * WV, 0, stream>>>(
      efeat, src, dst, Weh, Woh, sproj, dproj, b1, b_o, gamma, beta, out_e);

  (void)hipMemcpyAsync(out_n, nfeat, (size_t)N_NODES * DIM * sizeof(float),
                       hipMemcpyDeviceToDevice, stream);
}
